// MixedGNN_34239479284395
// MI455X (gfx1250) — matrix operand order repaired and checked by reading, not compiled, not run on hardware
//
#include <hip/hip_runtime.h>
#include <hip/hip_bf16.h>
#include <cmath>
#include <stdint.h>

// ---------------- types for WMMA ----------------
typedef __attribute__((ext_vector_type(16))) __bf16 v16bf;
typedef __attribute__((ext_vector_type(8)))  float  v8f;

#define D_H 128
#define H_HEADS 2
#define D_OUT 32
#define NEG_SLOPE 0.2f

__device__ __forceinline__ unsigned short f2bf(float f) {
  unsigned int u = __float_as_uint(f);
  u += 0x7FFFu + ((u >> 16) & 1u);   // round-to-nearest-even
  return (unsigned short)(u >> 16);
}

// ---------------- tiny utility kernels ----------------
__global__ void k_fill(float* __restrict__ p, long long n, float v) {
  long long i = (long long)blockIdx.x * blockDim.x + threadIdx.x;
  if (i < n) p[i] = v;
}

__global__ void k_f32_to_bf16(const float* __restrict__ in, unsigned short* __restrict__ out, long long n) {
  long long i = (long long)blockIdx.x * blockDim.x + threadIdx.x;
  if (i < n) out[i] = f2bf(in[i]);
}

// W[K x Kout] fp32 -> Wt[Kout x K] bf16 (so B fragments load contiguously)
__global__ void k_transpose_bf16(const float* __restrict__ W, unsigned short* __restrict__ Wt,
                                 int K, int Kout) {
  int i = blockIdx.x * blockDim.x + threadIdx.x;
  if (i >= K * Kout) return;
  int k = i / Kout, c = i % Kout;
  Wt[(long long)c * K + k] = f2bf(W[i]);
}

// ---------------- WMMA GEMM: C[N x Kout] = A[N x K]*B + (A2*B2) + bias, opt ReLU ----------------
// A, A2: row-major bf16 [N x K]; Bt, B2t: bf16 [Kout x K] (B transposed). K multiple of 32.
// One wave (32 lanes) computes one 16x16 tile.  grid = (ceil(N/16), Kout/16), block = 32.
__global__ void k_wmma_gemm(const unsigned short* __restrict__ A,
                            const unsigned short* __restrict__ Bt,
                            const unsigned short* __restrict__ A2,
                            const unsigned short* __restrict__ B2t,
                            const float* __restrict__ bias,
                            float* __restrict__ C,
                            unsigned short* __restrict__ Cbf,
                            int Nrows, int K, int Kout, int relu) {
  const int lane = threadIdx.x & 31;
  const int m    = lane & 15;     // A row within tile / B column within tile
  const int hh   = lane >> 4;     // half select
  const long long rowBase = (long long)blockIdx.x * 16;
  const int       colBase = blockIdx.y * 16;

  long long ar = rowBase + m; if (ar >= Nrows) ar = Nrows - 1;   // clamp (keeps EXEC uniform)
  const unsigned short* aRow = A  + ar * K;
  const unsigned short* bRow = Bt + (long long)(colBase + m) * K;

  v8f acc = {};
  for (int k0 = 0; k0 < K; k0 += 32) {
    v16bf af, bv;
    // A 16x32 bf16 fragment: lane half hh -> K = k0+hh*8+[0..7] and k0+16+hh*8+[0..7]
    ((uint4*)&af)[0] = *(const uint4*)(aRow + k0 + hh * 8);
    ((uint4*)&af)[1] = *(const uint4*)(aRow + k0 + 16 + hh * 8);
    // B 32x16 bf16 fragment: lane half hh -> K = k0+hh*8+[0..7] and k0+16+hh*8+[0..7]
    ((uint4*)&bv)[0] = *(const uint4*)(bRow + k0 + hh * 8);
    ((uint4*)&bv)[1] = *(const uint4*)(bRow + k0 + 16 + hh * 8);
    acc = __builtin_amdgcn_wmma_f32_16x16x32_bf16(false, af, false, bv, (short)0, acc, false, false);
  }
  if (A2 != nullptr) {
    const unsigned short* aRow2 = A2  + ar * K;
    const unsigned short* bRow2 = B2t + (long long)(colBase + m) * K;
    for (int k0 = 0; k0 < K; k0 += 32) {
      v16bf af, bv;
      ((uint4*)&af)[0] = *(const uint4*)(aRow2 + k0 + hh * 8);
      ((uint4*)&af)[1] = *(const uint4*)(aRow2 + k0 + 16 + hh * 8);
      ((uint4*)&bv)[0] = *(const uint4*)(bRow2 + k0 + hh * 8);
      ((uint4*)&bv)[1] = *(const uint4*)(bRow2 + k0 + 16 + hh * 8);
      acc = __builtin_amdgcn_wmma_f32_16x16x32_bf16(false, af, false, bv, (short)0, acc, false, false);
    }
  }
  // C/D layout: lane n = lane&15, VGPR v -> row v + 8*hh
  const int col = colBase + m;
  const float bval = bias ? bias[col] : 0.0f;
#pragma unroll
  for (int v = 0; v < 8; ++v) {
    long long row = rowBase + v + hh * 8;
    if (row < Nrows) {
      float val = acc[v] + bval;
      if (relu) val = fmaxf(val, 0.0f);
      C[row * Kout + col] = val;
      if (Cbf) Cbf[row * Kout + col] = f2bf(val);
    }
  }
}

// ---------------- GCN ----------------
__global__ void k_deg(const int* __restrict__ dst, int E, float* __restrict__ deg) {
  int e = blockIdx.x * blockDim.x + threadIdx.x;
  if (e < E) atomicAdd(&deg[dst[e]], 1.0f);
}

__global__ void k_dinv(const float* __restrict__ deg, float* __restrict__ dinv, int n) {
  int i = blockIdx.x * blockDim.x + threadIdx.x;
  if (i < n) dinv[i] = rsqrtf(fmaxf(deg[i], 1e-12f));
}

// self-loop contribution initializes the accumulator (runs before edge scatter)
__global__ void k_gcn_self(const float* __restrict__ xw, const float* __restrict__ dinv,
                           float* __restrict__ out) {
  long long n = blockIdx.x; int f = threadIdx.x;
  float di = dinv[n];
  out[n * D_H + f] = xw[n * D_H + f] * di * di;
}

__global__ void k_gcn_scatter(const int* __restrict__ src, const int* __restrict__ dst,
                              const float* __restrict__ xw, const float* __restrict__ dinv,
                              float* __restrict__ out) {
  long long e = blockIdx.x; int f = threadIdx.x;
  int s = src[e], d = dst[e];
  float norm = dinv[s] * dinv[d];
  atomicAdd(&out[(long long)d * D_H + f], xw[(long long)s * D_H + f] * norm);
}

__global__ void k_bias_relu_conv(const float* __restrict__ acc, const float* __restrict__ bias,
                                 float* __restrict__ outf, unsigned short* __restrict__ outbf) {
  long long n = blockIdx.x; int f = threadIdx.x;
  float v = fmaxf(acc[n * D_H + f] + bias[f], 0.0f);
  outf[n * D_H + f] = v;
  if (outbf) outbf[n * D_H + f] = f2bf(v);
}

// ---------------- GAT ----------------
// one wave per (node, head): 128-dim dot products, wave32 shuffle reduce
__global__ void k_gat_scores(const float* __restrict__ hgat, const float* __restrict__ att_src,
                             const float* __restrict__ att_dst,
                             float* __restrict__ a_s, float* __restrict__ a_d, int NH) {
  int gw   = (blockIdx.x * blockDim.x + threadIdx.x) >> 5;
  int lane = threadIdx.x & 31;
  if (gw >= NH) return;
  int node = gw >> 1, head = gw & 1;
  const float* hp = hgat + (long long)node * (H_HEADS * D_H) + head * D_H;
  const float* as = att_src + head * D_H;
  const float* ad = att_dst + head * D_H;
  float ss = 0.f, sd = 0.f;
#pragma unroll
  for (int j = 0; j < 4; ++j) {
    float hv = hp[lane + 32 * j];
    ss += hv * as[lane + 32 * j];
    sd += hv * ad[lane + 32 * j];
  }
  for (int o = 16; o > 0; o >>= 1) { ss += __shfl_xor(ss, o, 32); sd += __shfl_xor(sd, o, 32); }
  if (lane == 0) { a_s[gw] = ss; a_d[gw] = sd; }
}

__device__ __forceinline__ void atomicMaxF(float* addr, float v) {
  if (v >= 0.f) atomicMax((int*)addr, __float_as_int(v));
  else          atomicMin((unsigned int*)addr, __float_as_uint(v));
}

// phase 1: leaky-relu score per (edge, head), segment max via atomic
__global__ void k_gat_edge1(const int* __restrict__ src, const int* __restrict__ dst,
                            const float* __restrict__ a_s, const float* __restrict__ a_d,
                            float* __restrict__ ev, float* __restrict__ mmax, int E, int N) {
  int i = blockIdx.x * blockDim.x + threadIdx.x;
  int total = (E + N) * H_HEADS;
  if (i >= total) return;
  int head = i & 1, e = i >> 1;
  int s = (e < E) ? src[e] : (e - E);
  int d = (e < E) ? dst[e] : (e - E);
  float x = a_s[s * H_HEADS + head] + a_d[d * H_HEADS + head];
  x = (x >= 0.f) ? x : NEG_SLOPE * x;
  ev[i] = x;
  atomicMaxF(&mmax[d * H_HEADS + head], x);
}

// phase 2: exp(e - max) and segment sum
__global__ void k_gat_edge2(const int* __restrict__ dst, const float* __restrict__ mmax,
                            float* __restrict__ ev, float* __restrict__ ssum, int E, int N) {
  int i = blockIdx.x * blockDim.x + threadIdx.x;
  int total = (E + N) * H_HEADS;
  if (i >= total) return;
  int head = i & 1, e = i >> 1;
  int d = (e < E) ? dst[e] : (e - E);
  float ex = __expf(ev[i] - mmax[d * H_HEADS + head]);
  ev[i] = ex;
  atomicAdd(&ssum[d * H_HEADS + head], ex);
}

// phase 3: scatter alpha-weighted messages; one 128-thread block per (edge, head)
__global__ void k_gat_scatter(const int* __restrict__ src, const int* __restrict__ dst,
                              const float* __restrict__ ev, const float* __restrict__ ssum,
                              const float* __restrict__ hgat, float* __restrict__ acc,
                              int E) {
  int i = blockIdx.x;
  int head = i & 1, e = i >> 1;
  int s = (e < E) ? src[e] : (e - E);
  int d = (e < E) ? dst[e] : (e - E);
  float alpha = ev[i] / ssum[d * H_HEADS + head];
  int f = threadIdx.x;
  atomicAdd(&acc[(long long)d * (H_HEADS * D_H) + head * D_H + f],
            hgat[(long long)s * (H_HEADS * D_H) + head * D_H + f] * alpha);
}

// mean over heads + bias + relu, fp32 + bf16 outputs
__global__ void k_gat_post(const float* __restrict__ acc, const float* __restrict__ bias,
                           float* __restrict__ outf, unsigned short* __restrict__ outbf) {
  long long n = blockIdx.x; int f = threadIdx.x;
  float v0 = acc[n * (H_HEADS * D_H) + f];
  float v1 = acc[n * (H_HEADS * D_H) + D_H + f];
  float v = fmaxf(0.5f * (v0 + v1) + bias[f], 0.0f);
  outf[n * D_H + f] = v;
  outbf[n * D_H + f] = f2bf(v);
}

// ---------------- SAGE ----------------
__global__ void k_sage_scatter(const int* __restrict__ src, const int* __restrict__ dst,
                               const float* __restrict__ h, float* __restrict__ agg) {
  long long e = blockIdx.x; int f = threadIdx.x;
  int s = src[e], d = dst[e];
  atomicAdd(&agg[(long long)d * D_H + f], h[(long long)s * D_H + f]);
}

__global__ void k_sage_div(const float* __restrict__ agg, const float* __restrict__ deg,
                           unsigned short* __restrict__ agg16) {
  long long n = blockIdx.x; int f = threadIdx.x;
  float v = agg[n * D_H + f] / fmaxf(deg[n], 1.0f);
  agg16[n * D_H + f] = f2bf(v);
}

// ---------------- log-softmax: one wave32 per row of 32 logits ----------------
__global__ void k_logsoftmax(const float* __restrict__ logits, float* __restrict__ out, int N) {
  int wid  = threadIdx.x >> 5;
  int lane = threadIdx.x & 31;
  long long row = (long long)blockIdx.x * 8 + wid;
  if (row >= N) return;
  float v = logits[row * D_OUT + lane];
  float mx = v;
  for (int o = 16; o > 0; o >>= 1) mx = fmaxf(mx, __shfl_xor(mx, o, 32));
  float ex = __expf(v - mx);
  float sm = ex;
  for (int o = 16; o > 0; o >>= 1) sm += __shfl_xor(sm, o, 32);
  out[row * D_OUT + lane] = v - mx - __logf(sm);
}

// ---------------- launcher ----------------
extern "C" void kernel_launch(void* const* d_in, const int* in_sizes, int n_in,
                              void* d_out, int out_size, void* d_ws, size_t ws_size,
                              hipStream_t stream) {
  const float* x        = (const float*)d_in[0];
  const float* W_gcn    = (const float*)d_in[1];
  const float* b_gcn    = (const float*)d_in[2];
  const float* W_gat    = (const float*)d_in[3];
  const float* att_src  = (const float*)d_in[4];
  const float* att_dst  = (const float*)d_in[5];
  const float* b_gat    = (const float*)d_in[6];
  const float* W_sage_l = (const float*)d_in[7];
  const float* b_sage_l = (const float*)d_in[8];
  const float* W_sage_r = (const float*)d_in[9];
  const float* W_out    = (const float*)d_in[10];
  const float* b_out    = (const float*)d_in[11];
  const int*   ei       = (const int*)d_in[12];

  const int N = in_sizes[0] / D_H;
  const int E = in_sizes[12] / 2;
  const int* src = ei;
  const int* dst = ei + E;

  // --- bump allocator on workspace ---
  char* base = (char*)d_ws;
  size_t off = 0;
  auto alloc = [&](size_t bytes) -> void* {
    void* p = base + off;
    off = (off + bytes + 255) & ~(size_t)255;
    return p;
  };
  unsigned short* x16     = (unsigned short*)alloc((size_t)N * D_H * 2);
  unsigned short* wgcn_t  = (unsigned short*)alloc((size_t)D_H * D_H * 2);
  unsigned short* wgat_t  = (unsigned short*)alloc((size_t)D_H * H_HEADS * D_H * 2);
  unsigned short* wsl_t   = (unsigned short*)alloc((size_t)D_H * D_H * 2);
  unsigned short* wsr_t   = (unsigned short*)alloc((size_t)D_H * D_H * 2);
  unsigned short* wout_t  = (unsigned short*)alloc((size_t)D_H * D_OUT * 2);
  float* xw      = (float*)alloc((size_t)N * D_H * 4);
  float* deg     = (float*)alloc((size_t)N * 4);
  float* dinv    = (float*)alloc((size_t)N * 4);
  float* gcnacc  = (float*)alloc((size_t)N * D_H * 4);
  float* h1f     = (float*)alloc((size_t)N * D_H * 4);
  unsigned short* h1_16 = (unsigned short*)alloc((size_t)N * D_H * 2);
  float* hgat    = (float*)alloc((size_t)N * H_HEADS * D_H * 4);
  float* a_s     = (float*)alloc((size_t)N * H_HEADS * 4);
  float* a_d     = (float*)alloc((size_t)N * H_HEADS * 4);
  float* mmax    = (float*)alloc((size_t)N * H_HEADS * 4);
  float* ssum    = (float*)alloc((size_t)N * H_HEADS * 4);
  float* ev      = (float*)alloc((size_t)(E + N) * H_HEADS * 4);
  float* gatacc  = (float*)alloc((size_t)N * H_HEADS * D_H * 4);
  float* h2f     = (float*)alloc((size_t)N * D_H * 4);
  unsigned short* h2_16 = (unsigned short*)alloc((size_t)N * D_H * 2);
  float* deg2    = (float*)alloc((size_t)N * 4);
  float* aggf    = (float*)alloc((size_t)N * D_H * 4);
  unsigned short* agg16 = (unsigned short*)alloc((size_t)N * D_H * 2);
  float* h3f     = (float*)alloc((size_t)N * D_H * 4);
  unsigned short* h3_16 = (unsigned short*)alloc((size_t)N * D_H * 2);
  float* logits  = (float*)alloc((size_t)N * D_OUT * 4);
  (void)ws_size; (void)n_in; (void)out_size;

  const int rowTiles = (N + 15) / 16;
  auto cdiv = [](long long a, long long b) { return (int)((a + b - 1) / b); };

  // --- prep: bf16 conversions / weight transposes ---
  k_f32_to_bf16<<<cdiv((long long)N * D_H, 256), 256, 0, stream>>>(x, x16, (long long)N * D_H);
  k_transpose_bf16<<<cdiv(D_H * D_H, 256), 256, 0, stream>>>(W_gcn, wgcn_t, D_H, D_H);
  k_transpose_bf16<<<cdiv(D_H * H_HEADS * D_H, 256), 256, 0, stream>>>(W_gat, wgat_t, D_H, H_HEADS * D_H);
  k_transpose_bf16<<<cdiv(D_H * D_H, 256), 256, 0, stream>>>(W_sage_l, wsl_t, D_H, D_H);
  k_transpose_bf16<<<cdiv(D_H * D_H, 256), 256, 0, stream>>>(W_sage_r, wsr_t, D_H, D_H);
  k_transpose_bf16<<<cdiv(D_H * D_OUT, 256), 256, 0, stream>>>(W_out, wout_t, D_H, D_OUT);

  // --- GCN: xw = x @ W_gcn (WMMA), symmetric-norm scatter, bias+relu ---
  k_wmma_gemm<<<dim3(rowTiles, D_H / 16), 32, 0, stream>>>(
      x16, wgcn_t, nullptr, nullptr, nullptr, xw, nullptr, N, D_H, D_H, 0);
  k_fill<<<cdiv(N, 256), 256, 0, stream>>>(deg, N, 1.0f);                 // self loop
  k_deg<<<cdiv(E, 256), 256, 0, stream>>>(dst, E, deg);
  k_dinv<<<cdiv(N, 256), 256, 0, stream>>>(deg, dinv, N);
  k_gcn_self<<<N, D_H, 0, stream>>>(xw, dinv, gcnacc);
  k_gcn_scatter<<<E, D_H, 0, stream>>>(src, dst, xw, dinv, gcnacc);
  k_bias_relu_conv<<<N, D_H, 0, stream>>>(gcnacc, b_gcn, h1f, h1_16);

  // --- GAT: hgat = h1 @ W_gat (WMMA), attention softmax, scatter, head-mean ---
  k_wmma_gemm<<<dim3(rowTiles, (H_HEADS * D_H) / 16), 32, 0, stream>>>(
      h1_16, wgat_t, nullptr, nullptr, nullptr, hgat, nullptr, N, D_H, H_HEADS * D_H, 0);
  k_gat_scores<<<cdiv((long long)N * H_HEADS * 32, 256), 256, 0, stream>>>(
      hgat, att_src, att_dst, a_s, a_d, N * H_HEADS);
  k_fill<<<cdiv((long long)N * H_HEADS, 256), 256, 0, stream>>>(mmax, (long long)N * H_HEADS, -INFINITY);
  k_fill<<<cdiv((long long)N * H_HEADS, 256), 256, 0, stream>>>(ssum, (long long)N * H_HEADS, 0.0f);
  const int EH = (E + N) * H_HEADS;
  k_gat_edge1<<<cdiv(EH, 256), 256, 0, stream>>>(src, dst, a_s, a_d, ev, mmax, E, N);
  k_gat_edge2<<<cdiv(EH, 256), 256, 0, stream>>>(dst, mmax, ev, ssum, E, N);
  k_fill<<<cdiv((long long)N * H_HEADS * D_H, 256), 256, 0, stream>>>(gatacc, (long long)N * H_HEADS * D_H, 0.0f);
  k_gat_scatter<<<EH, D_H, 0, stream>>>(src, dst, ev, ssum, hgat, gatacc, E);
  k_gat_post<<<N, D_H, 0, stream>>>(gatacc, b_gat, h2f, h2_16);

  // --- SAGE: mean aggregate, fused dual-WMMA GEMM (agg@Wl + h2@Wr) + bias + relu ---
  k_fill<<<cdiv(N, 256), 256, 0, stream>>>(deg2, N, 0.0f);
  k_deg<<<cdiv(E, 256), 256, 0, stream>>>(dst, E, deg2);
  k_fill<<<cdiv((long long)N * D_H, 256), 256, 0, stream>>>(aggf, (long long)N * D_H, 0.0f);
  k_sage_scatter<<<E, D_H, 0, stream>>>(src, dst, h2f, aggf);
  k_sage_div<<<N, D_H, 0, stream>>>(aggf, deg2, agg16);
  k_wmma_gemm<<<dim3(rowTiles, D_H / 16), 32, 0, stream>>>(
      agg16, wsl_t, h2_16, wsr_t, b_sage_l, h3f, h3_16, N, D_H, D_H, 1);

  // --- output layer (WMMA) + wave32 log-softmax ---
  k_wmma_gemm<<<dim3(rowTiles, D_OUT / 16), 32, 0, stream>>>(
      h3_16, wout_t, nullptr, nullptr, b_out, logits, nullptr, N, D_H, D_OUT, 0);
  k_logsoftmax<<<cdiv(N, 8), 256, 0, stream>>>(logits, (float*)d_out, N);
}
